// net_2310692405831
// MI455X (gfx1250) — matrix operand order repaired and checked by reading, not compiled, not run on hardware
//
#include <hip/hip_runtime.h>

typedef __attribute__((ext_vector_type(16))) __bf16 bf16x16;
typedef __attribute__((ext_vector_type(8)))  __bf16 bf16x8;
typedef __attribute__((ext_vector_type(8)))  float  f32x8;

#define BATCH   32
#define TOK     197     // real tokens
#define TOKP    224     // padded tokens (multiple of 32)
#define NPATCH  196
#define DMODEL  512
#define PDIM    768
#define NHEAD   8
#define HDIM    64
#define MLPD    2048
#define NL      10
#define NCLSS   10
#define LOGLD   16      // padded logit row
#define ROWSP   (BATCH*TOKP)   // 7168 (multiple of 16)
#define EROWS   (BATCH*NPATCH) // 6272 (multiple of 16)

// ---------- small device helpers ----------
__device__ __forceinline__ unsigned short f2bf(float f) {
  unsigned int u = __float_as_uint(f);
  u += 0x7fffu + ((u >> 16) & 1u);      // RNE
  return (unsigned short)(u >> 16);
}
__device__ __forceinline__ float bf2f(unsigned short u) {
  return __uint_as_float(((unsigned int)u) << 16);
}
__device__ __forceinline__ float gelu_f(float x) {
  return 0.5f * x * (1.0f + erff(x * 0.70710678118654752f));
}
__device__ __forceinline__ bf16x16 load_a_frag(const unsigned short* p) {
  // ISA A layout: lane-group g holds k = g*8 + j (j<8) and k = g*8 + 8 + j (j>=8)
  bf16x8 lo = *(const bf16x8*)p;          // 16B aligned
  bf16x8 hi = *(const bf16x8*)(p + 16);   // +32B
  return __builtin_shufflevector(lo, hi, 0,1,2,3,4,5,6,7,8,9,10,11,12,13,14,15);
}

// ---------- batched WMMA GEMM: C[M,N] = alpha*A[M,K]*B^T + epilogue ----------
// A bf16 (M,K) row-major lda; B bf16 (N,K) row-major ldb ("weights" layout).
// Batch offsets: i1=batch/bdiv, i2=batch%bdiv -> ptr += i1*s?1 + i2*s?2.
// mode: 0=store 1=+bias 2=+bias,gelu 3=+bias then C_f32 += v
// Requires: M%16==0, N%(16*TN)==0, K%32==0, all reads in-bounds (caller pads).
template<int TN>
__global__ void __launch_bounds__(256)
wmma_gemm_bf16(const unsigned short* __restrict__ A, long long sA1, long long sA2, int lda,
               const unsigned short* __restrict__ B, long long sB1, long long sB2, int ldb,
               const float* __restrict__ bias,
               void* __restrict__ C, long long sC1, long long sC2, int ldc,
               int M, int N, int K, int bdiv, int mode, int out_bf16, float alpha) {
  int batch = blockIdx.y;
  int i1 = batch / bdiv, i2 = batch % bdiv;
  const unsigned short* Ab = A + i1 * sA1 + i2 * sA2;
  const unsigned short* Bb = B + i1 * sB1 + i2 * sB2;
  long long coff = i1 * sC1 + i2 * sC2;

  int Mt = M >> 4, Ng = N / (16 * TN);
  int wave = blockIdx.x * (blockDim.x >> 5) + (threadIdx.x >> 5);
  if (wave >= Mt * Ng) return;
  int mt = wave / Ng, ng = wave % Ng;
  int m0 = mt << 4, n0 = ng * (16 * TN);
  int lane = threadIdx.x & 31;
  int g = lane >> 4, r = lane & 15;

  const unsigned short* arow = Ab + (long long)(m0 + r) * lda + g * 8;
  const unsigned short* brow[TN];
#pragma unroll
  for (int t = 0; t < TN; ++t)
    brow[t] = Bb + (long long)(n0 + t * 16 + r) * ldb + g * 8;

  f32x8 acc[TN];
#pragma unroll
  for (int t = 0; t < TN; ++t) acc[t] = (f32x8){0.f,0.f,0.f,0.f,0.f,0.f,0.f,0.f};

  for (int k0 = 0; k0 < K; k0 += 32) {
    bf16x16 av = load_a_frag(arow + k0);
#pragma unroll
    for (int t = 0; t < TN; ++t) {
      bf16x16 bv = load_a_frag(brow[t] + k0);
      acc[t] = __builtin_amdgcn_wmma_f32_16x16x32_bf16(false, av, false, bv,
                                                       (short)0, acc[t], false, false);
    }
  }

#pragma unroll
  for (int t = 0; t < TN; ++t) {
#pragma unroll
    for (int i = 0; i < 8; ++i) {
      int m = m0 + g * 8 + i;
      int n = n0 + t * 16 + r;
      float v = acc[t][i] * alpha;
      if (mode >= 1) v += bias[n];
      if (mode == 2) v = gelu_f(v);
      long long idx = coff + (long long)m * ldc + n;
      if (mode == 3)     ((float*)C)[idx] += v;
      else if (out_bf16) ((unsigned short*)C)[idx] = f2bf(v);
      else               ((float*)C)[idx] = v;
    }
  }
}

// ---------- f32 -> bf16 ----------
__global__ void cvt_f32_bf16(const float* __restrict__ in, unsigned short* __restrict__ out, int n) {
  int i = blockIdx.x * 256 + threadIdx.x;
  if (i < n) out[i] = f2bf(in[i]);
}

// ---------- Wc2 -> bf16 padded to 16 rows; bc2 -> padded f32 ----------
__global__ void pad_head_kernel(const float* __restrict__ Wc2, const float* __restrict__ bc2,
                                unsigned short* __restrict__ Wc2p, float* __restrict__ bc2p) {
  int i = blockIdx.x * 256 + threadIdx.x;
  if (i < LOGLD * MLPD) {
    int row = i / MLPD;
    Wc2p[i] = (row < NCLSS) ? f2bf(Wc2[i - (row - (row < NCLSS ? row : 0)) * 0 + 0*0]) : 0;
  }
  if (i < LOGLD * MLPD) {
    int row = i / MLPD, col = i % MLPD;
    Wc2p[i] = (row < NCLSS) ? f2bf(Wc2[row * MLPD + col]) : (unsigned short)0;
  }
  if (i < LOGLD) bc2p[i] = (i < NCLSS) ? bc2[i] : 0.0f;
}

// ---------- im2col (bf16) ----------
__global__ void im2col_kernel(const float* __restrict__ x, unsigned short* __restrict__ xp) {
  int i = blockIdx.x * 256 + threadIdx.x;
  if (i >= EROWS * PDIM) return;
  int k = i % PDIM;
  int n = (i / PDIM) % NPATCH;
  int bi = i / (PDIM * NPATCH);
  int c = k % 3, pc = (k / 3) % 16, pr = k / 48;
  int px = n % 14, py = n / 14;
  int row = py * 16 + pr, col = px * 16 + pc;
  xp[i] = f2bf(x[(((long long)bi * 3 + c) * 224 + row) * 224 + col]);
}

// ---------- assemble padded residual stream Z from E + cls + pos ----------
__global__ void pos_cls_kernel(float* __restrict__ Z, const float* __restrict__ E,
                               const float* __restrict__ pos, const float* __restrict__ cls,
                               const float* __restrict__ bp) {
  int i = blockIdx.x * 256 + threadIdx.x;
  if (i >= ROWSP * DMODEL) return;
  int d = i % DMODEL;
  int t = (i / DMODEL) % TOKP;
  int bi = i / (DMODEL * TOKP);
  float v;
  if (t == 0)        v = cls[d] + pos[d];
  else if (t < TOK)  v = E[((long long)bi * NPATCH + (t - 1)) * DMODEL + d] + bp[d] + pos[t * DMODEL + d];
  else               v = 0.0f;
  Z[i] = v;
}

// ---------- LayerNorm over 512 -> bf16 (1 block/row, 256 thr) ----------
__global__ void ln_bf16_kernel(const float* __restrict__ in, const float* __restrict__ g,
                               const float* __restrict__ b, unsigned short* __restrict__ out) {
  int row = blockIdx.x, t = threadIdx.x;
  __shared__ float red[256];
  const float* x = in + (long long)row * DMODEL;
  float v0 = x[t], v1 = x[t + 256];
  red[t] = v0 + v1; __syncthreads();
  for (int o = 128; o > 0; o >>= 1) { if (t < o) red[t] += red[t + o]; __syncthreads(); }
  float mu = red[0] * (1.0f / DMODEL); __syncthreads();
  float d0 = v0 - mu, d1 = v1 - mu;
  red[t] = d0 * d0 + d1 * d1; __syncthreads();
  for (int o = 128; o > 0; o >>= 1) { if (t < o) red[t] += red[t + o]; __syncthreads(); }
  float rinv = rsqrtf(red[0] * (1.0f / DMODEL) + 1e-5f);
  unsigned short* o0 = out + (long long)row * DMODEL;
  o0[t]       = f2bf(d0 * rinv * g[t] + b[t]);
  o0[t + 256] = f2bf(d1 * rinv * g[t + 256] + b[t + 256]);
}

// ---------- softmax over 197, zero pad to 224, bf16 out ----------
__global__ void softmax_bf16_kernel(const float* __restrict__ S, unsigned short* __restrict__ P) {
  int row = blockIdx.x, t = threadIdx.x;
  __shared__ float red[256];
  float v = (t < TOK) ? S[(long long)row * TOKP + t] : -3.4e38f;
  red[t] = v; __syncthreads();
  for (int o = 128; o > 0; o >>= 1) { if (t < o) red[t] = fmaxf(red[t], red[t + o]); __syncthreads(); }
  float mx = red[0]; __syncthreads();
  float e = (t < TOK) ? expf(v - mx) : 0.0f;
  red[t] = e; __syncthreads();
  for (int o = 128; o > 0; o >>= 1) { if (t < o) red[t] += red[t + o]; __syncthreads(); }
  float inv = 1.0f / red[0];
  if (t < TOKP) P[(long long)row * TOKP + t] = (t < TOK) ? f2bf(e * inv) : (unsigned short)0;
}

// ---------- transpose V: (bh) x (64 x 224) from padded QKV ----------
__global__ void vtrans_kernel(const unsigned short* __restrict__ qkv, unsigned short* __restrict__ VT) {
  int i = blockIdx.x * 256 + threadIdx.x;
  if (i >= BATCH * NHEAD * HDIM * TOKP) return;
  int t = i % TOKP;
  int d = (i / TOKP) % HDIM;
  int bh = i / (TOKP * HDIM);
  int bi = bh >> 3, h = bh & 7;
  VT[i] = qkv[((long long)bi * TOKP + t) * (3 * DMODEL) + 2 * DMODEL + h * HDIM + d];
}

// ---------- halt row dot (2048) + bias + sigmoid ----------
__global__ void rowdot_sigmoid_kernel(const unsigned short* __restrict__ Hbf,
                                      const unsigned short* __restrict__ w,
                                      const float* __restrict__ bias, float* __restrict__ svec) {
  int row = blockIdx.x, t = threadIdx.x;
  __shared__ float red[256];
  float acc = 0.0f;
  for (int j = t; j < MLPD; j += 256)
    acc += bf2f(Hbf[(long long)row * MLPD + j]) * bf2f(w[j]);
  red[t] = acc; __syncthreads();
  for (int o = 128; o > 0; o >>= 1) { if (t < o) red[t] += red[t + o]; __syncthreads(); }
  if (t == 0) { float s = red[0] + bias[0]; svec[row] = 1.0f / (1.0f + expf(-s)); }
}

// ---------- mean of sigmoid over 197 real tokens ----------
__global__ void halt_mean_kernel(const float* __restrict__ svec, float* __restrict__ halts, int layer) {
  int bi = blockIdx.x, t = threadIdx.x;
  __shared__ float red[256];
  red[t] = (t < TOK) ? svec[bi * TOKP + t] : 0.0f; __syncthreads();
  for (int o = 128; o > 0; o >>= 1) { if (t < o) red[t] += red[t + o]; __syncthreads(); }
  if (t == 0) halts[bi * NL + layer] = red[0] * (1.0f / TOK);
}

// ---------- record cls token ----------
__global__ void copy_cls_kernel(const float* __restrict__ Z, float* __restrict__ clso, int layer) {
  int bi = blockIdx.x, d = threadIdx.x; // 512 threads
  clso[((long long)bi * NL + layer) * DMODEL + d] = Z[(long long)bi * TOKP * DMODEL + d];
}

// ---------- final loss (one wave) ----------
__global__ void loss_kernel(const float* __restrict__ logits, const float* __restrict__ halts,
                            const int* __restrict__ y, float* __restrict__ out) {
  int bi = threadIdx.x; // 0..31
  __shared__ float acc[32];
  float ce[NL];
  int lab = y[bi];
  for (int l = 0; l < NL; ++l) {
    const float* lg = logits + ((long long)bi * NL + l) * LOGLD;
    float mx = lg[0];
    for (int c = 1; c < NCLSS; ++c) mx = fmaxf(mx, lg[c]);
    float s = 0.0f;
    for (int c = 0; c < NCLSS; ++c) s += expf(lg[c] - mx);
    ce[l] = (mx + logf(s)) - lg[lab];
  }
  int idx = 0; float best = ce[0];
  for (int l = 1; l < NL; ++l) if (ce[l] < best) { best = ce[l]; idx = l; }
  idx += 1;
  float denom = (float)idx, ces = 0.0f, bces = 0.0f;
  for (int l = 0; l < NL; ++l) {
    if (l < idx) {
      ces += ce[l];
      float hl = (l == idx - 1) ? 0.0f : 1.0f;
      float p = fminf(fmaxf(halts[bi * NL + l], 1e-7f), 1.0f - 1e-7f);
      bces += -(hl * logf(p) + (1.0f - hl) * log1pf(-p));
    }
  }
  acc[bi] = (ces + bces) / denom;
  __syncthreads();
  if (bi == 0) {
    float s = 0.0f;
    for (int i = 0; i < 32; ++i) s += acc[i];
    out[0] = s * (1.0f / 32.0f);
  }
}

// ---------- host-side GEMM launcher (TN dispatch) ----------
static inline void launch_gemm(hipStream_t st,
    const unsigned short* A, long long sA1, long long sA2, int lda,
    const unsigned short* B, long long sB1, long long sB2, int ldb,
    const float* bias, void* C, long long sC1, long long sC2, int ldc,
    int M, int N, int K, int nbatch, int bdiv, int mode, int out_bf16, float alpha) {
  int Mt = M / 16;
  if (N % 64 == 0) {
    int gx = (Mt * (N / 64) + 7) / 8;
    wmma_gemm_bf16<4><<<dim3(gx, nbatch), 256, 0, st>>>(A,sA1,sA2,lda,B,sB1,sB2,ldb,bias,
                                                        C,sC1,sC2,ldc,M,N,K,bdiv,mode,out_bf16,alpha);
  } else if (N % 32 == 0) {
    int gx = (Mt * (N / 32) + 7) / 8;
    wmma_gemm_bf16<2><<<dim3(gx, nbatch), 256, 0, st>>>(A,sA1,sA2,lda,B,sB1,sB2,ldb,bias,
                                                        C,sC1,sC2,ldc,M,N,K,bdiv,mode,out_bf16,alpha);
  } else {
    int gx = (Mt * (N / 16) + 7) / 8;
    wmma_gemm_bf16<1><<<dim3(gx, nbatch), 256, 0, st>>>(A,sA1,sA2,lda,B,sB1,sB2,ldb,bias,
                                                        C,sC1,sC2,ldc,M,N,K,bdiv,mode,out_bf16,alpha);
  }
}

extern "C" void kernel_launch(void* const* d_in, const int* in_sizes, int n_in,
                              void* d_out, int out_size, void* d_ws, size_t ws_size,
                              hipStream_t stream) {
  const float* x      = (const float*)d_in[0];
  const int*   y      = (const int*)  d_in[1];
  const float* pos    = (const float*)d_in[2];
  const float* cls    = (const float*)d_in[3];
  const float* Wp     = (const float*)d_in[4];
  const float* bp     = (const float*)d_in[5];
  const float* Wqkv   = (const float*)d_in[6];
  const float* Wo     = (const float*)d_in[7];
  const float* bo     = (const float*)d_in[8];
  const float* ln1_g  = (const float*)d_in[9];
  const float* ln1_b  = (const float*)d_in[10];
  const float* ln2_g  = (const float*)d_in[11];
  const float* ln2_b  = (const float*)d_in[12];
  const float* W1     = (const float*)d_in[13];
  const float* b1     = (const float*)d_in[14];
  const float* W2     = (const float*)d_in[15];
  const float* b2     = (const float*)d_in[16];
  const float* lnh_g  = (const float*)d_in[17];
  const float* lnh_b  = (const float*)d_in[18];
  const float* Wh1    = (const float*)d_in[19];
  const float* bh1    = (const float*)d_in[20];
  const float* Wh2    = (const float*)d_in[21];
  const float* bh2    = (const float*)d_in[22];
  const float* lnc_g  = (const float*)d_in[23];
  const float* lnc_b  = (const float*)d_in[24];
  const float* Wc1    = (const float*)d_in[25];
  const float* bc1    = (const float*)d_in[26];
  const float* Wc2    = (const float*)d_in[27];
  const float* bc2    = (const float*)d_in[28];

  // -------- workspace layout --------
  char* base = (char*)d_ws;
  size_t off = 0;
  auto take = [&](size_t bytes) { size_t o = off; off = (off + bytes + 255) & ~(size_t)255; return o; };
  const long long SCT = (long long)BATCH * NHEAD * TOKP * TOKP; // 12,845,056

  float*          Z      = (float*)         (base + take((size_t)ROWSP * DMODEL * 4));
  unsigned short* YBF    = (unsigned short*)(base + take((size_t)ROWSP * DMODEL * 2));
  unsigned short* QKVBF  = (unsigned short*)(base + take((size_t)ROWSP * 3 * DMODEL * 2));
  unsigned short* PROBS  = (unsigned short*)(base + take((size_t)SCT * 2)); // alias: im2col staging
  unsigned short* OBUF   = (unsigned short*)(base + take((size_t)ROWSP * DMODEL * 2));
  unsigned short* VT     = (unsigned short*)(base + take((size_t)BATCH * NHEAD * HDIM * TOKP * 2));
  // BIG: scores f32 (51.4MB) aliased with MLP hidden bf16 (29.4MB) and patch-embed E f32 (12.8MB)
  char*           BIG    =                   base + take((size_t)SCT * 4);
  float*          SCORES = (float*)BIG;
  unsigned short* HBF    = (unsigned short*)BIG;
  float*          EBUF   = (float*)BIG;
  float*          SVEC   = (float*)         (base + take((size_t)ROWSP * 4));
  float*          HALTS  = (float*)         (base + take((size_t)BATCH * NL * 4));
  float*          CLSO   = (float*)         (base + take((size_t)BATCH * NL * DMODEL * 4));
  unsigned short* YBF2   = (unsigned short*)(base + take((size_t)BATCH * NL * DMODEL * 2));
  unsigned short* HCBF   = (unsigned short*)(base + take((size_t)BATCH * NL * MLPD * 2));
  float*          LOGITS = (float*)         (base + take((size_t)BATCH * NL * LOGLD * 4));
  unsigned short* WPb    = (unsigned short*)(base + take((size_t)DMODEL * PDIM * 2));
  unsigned short* WQKVb  = (unsigned short*)(base + take((size_t)3 * DMODEL * DMODEL * 2));
  unsigned short* WOb    = (unsigned short*)(base + take((size_t)DMODEL * DMODEL * 2));
  unsigned short* W1b    = (unsigned short*)(base + take((size_t)MLPD * DMODEL * 2));
  unsigned short* W2b    = (unsigned short*)(base + take((size_t)DMODEL * MLPD * 2));
  unsigned short* WH1b   = (unsigned short*)(base + take((size_t)MLPD * DMODEL * 2));
  unsigned short* WH2b   = (unsigned short*)(base + take((size_t)MLPD * 2));
  unsigned short* WC1b   = (unsigned short*)(base + take((size_t)MLPD * DMODEL * 2));
  unsigned short* WC2b   = (unsigned short*)(base + take((size_t)LOGLD * MLPD * 2));
  float*          BC2P   = (float*)         (base + take((size_t)LOGLD * 4));
  unsigned short* XPBF   = PROBS; // alias (6272*768 bf16 fits in PROBS)

  auto cvt = [&](const float* s, unsigned short* d, int n) {
    cvt_f32_bf16<<<(n + 255) / 256, 256, 0, stream>>>(s, d, n);
  };
  cvt(Wp,   WPb,   DMODEL * PDIM);
  cvt(Wqkv, WQKVb, 3 * DMODEL * DMODEL);
  cvt(Wo,   WOb,   DMODEL * DMODEL);
  cvt(W1,   W1b,   MLPD * DMODEL);
  cvt(W2,   W2b,   DMODEL * MLPD);
  cvt(Wh1,  WH1b,  MLPD * DMODEL);
  cvt(Wh2,  WH2b,  MLPD);
  cvt(Wc1,  WC1b,  MLPD * DMODEL);
  pad_head_kernel<<<(LOGLD * MLPD + 255) / 256, 256, 0, stream>>>(Wc2, bc2, WC2b, BC2P);

  // -------- patch embed --------
  {
    int n = EROWS * PDIM;
    im2col_kernel<<<(n + 255) / 256, 256, 0, stream>>>(x, XPBF);
    launch_gemm(stream, XPBF, 0, 0, PDIM, WPb, 0, 0, PDIM, nullptr,
                EBUF, 0, 0, DMODEL, EROWS, DMODEL, PDIM, 1, 1, 0, 0, 1.0f);
    int m = ROWSP * DMODEL;
    pos_cls_kernel<<<(m + 255) / 256, 256, 0, stream>>>(Z, EBUF, pos, cls, bp);
  }

  const float scale = 0.04419417382415922f; // 512^-0.5

  for (int layer = 0; layer < NL; ++layer) {
    // ---- attention ----
    ln_bf16_kernel<<<ROWSP, 256, 0, stream>>>(Z, ln1_g, ln1_b, YBF);
    launch_gemm(stream, YBF, 0, 0, DMODEL, WQKVb, 0, 0, DMODEL, nullptr,
                QKVBF, 0, 0, 3 * DMODEL, ROWSP, 3 * DMODEL, DMODEL, 1, 1, 0, 1, 1.0f);
    // scores[b,h] = scale * Q K^T  (M=N=224, K=64)
    launch_gemm(stream, QKVBF, (long long)TOKP * 3 * DMODEL, HDIM, 3 * DMODEL,
                QKVBF + DMODEL, (long long)TOKP * 3 * DMODEL, HDIM, 3 * DMODEL,
                nullptr, SCORES, (long long)NHEAD * TOKP * TOKP, (long long)TOKP * TOKP, TOKP,
                TOKP, TOKP, HDIM, BATCH * NHEAD, NHEAD, 0, 0, scale);
    softmax_bf16_kernel<<<BATCH * NHEAD * TOKP, 256, 0, stream>>>(SCORES, PROBS);
    {
      int n = BATCH * NHEAD * HDIM * TOKP;
      vtrans_kernel<<<(n + 255) / 256, 256, 0, stream>>>(QKVBF, VT);
    }
    // o[b,h] = P V  (M=224, N=64, K=224; B = VT as (N,K))
    launch_gemm(stream, PROBS, (long long)NHEAD * TOKP * TOKP, (long long)TOKP * TOKP, TOKP,
                VT, (long long)NHEAD * HDIM * TOKP, (long long)HDIM * TOKP, TOKP,
                nullptr, OBUF, (long long)TOKP * DMODEL, HDIM, DMODEL,
                TOKP, HDIM, TOKP, BATCH * NHEAD, NHEAD, 0, 1, 1.0f);
    // z += o @ Wo^T + bo
    launch_gemm(stream, OBUF, 0, 0, DMODEL, WOb, 0, 0, DMODEL, bo,
                Z, 0, 0, DMODEL, ROWSP, DMODEL, DMODEL, 1, 1, 3, 0, 1.0f);
    // ---- mlp ----
    ln_bf16_kernel<<<ROWSP, 256, 0, stream>>>(Z, ln2_g, ln2_b, YBF);
    launch_gemm(stream, YBF, 0, 0, DMODEL, W1b, 0, 0, DMODEL, b1,
                HBF, 0, 0, MLPD, ROWSP, MLPD, DMODEL, 1, 1, 2, 1, 1.0f);
    launch_gemm(stream, HBF, 0, 0, MLPD, W2b, 0, 0, MLPD, b2,
                Z, 0, 0, DMODEL, ROWSP, DMODEL, MLPD, 1, 1, 3, 0, 1.0f);
    // ---- halting head ----
    ln_bf16_kernel<<<ROWSP, 256, 0, stream>>>(Z, lnh_g, lnh_b, YBF);
    launch_gemm(stream, YBF, 0, 0, DMODEL, WH1b, 0, 0, DMODEL, bh1,
                HBF, 0, 0, MLPD, ROWSP, MLPD, DMODEL, 1, 1, 2, 1, 1.0f);
    rowdot_sigmoid_kernel<<<ROWSP, 256, 0, stream>>>(HBF, WH2b, bh2, SVEC);
    halt_mean_kernel<<<BATCH, 256, 0, stream>>>(SVEC, HALTS, layer);
    copy_cls_kernel<<<BATCH, 512, 0, stream>>>(Z, CLSO, layer);
  }

  // -------- classification head --------
  ln_bf16_kernel<<<BATCH * NL, 256, 0, stream>>>(CLSO, lnc_g, lnc_b, YBF2);
  launch_gemm(stream, YBF2, 0, 0, DMODEL, WC1b, 0, 0, DMODEL, bc1,
              HCBF, 0, 0, MLPD, BATCH * NL, MLPD, DMODEL, 1, 1, 2, 1, 1.0f);
  launch_gemm(stream, HCBF, 0, 0, MLPD, WC2b, 0, 0, MLPD, BC2P,
              LOGITS, 0, 0, LOGLD, BATCH * NL, LOGLD, MLPD, 1, 1, 1, 0, 1.0f);

  loss_kernel<<<1, 32, 0, stream>>>(LOGITS, HALTS, y, (float*)d_out);
}
